// MSDeformAttn_31988916420841
// MI455X (gfx1250) — matrix operand order repaired and checked by reading, not compiled, not run on hardware
//
#include <hip/hip_runtime.h>

typedef float    v4f  __attribute__((ext_vector_type(4)));
typedef float    v8f  __attribute__((ext_vector_type(8)));
typedef _Float16 v4h  __attribute__((ext_vector_type(4)));
typedef _Float16 v8h  __attribute__((ext_vector_type(8)));
typedef _Float16 v16h __attribute__((ext_vector_type(16)));

#define NB   4
#define LQ   13294
#define LEN  13294
#define NTOT (NB * LQ)   /* 53176 rows */
#define MH   8

// ---------------------------------------------------------------------------
// Elementwise fp32 -> f16 (float4 in, half4 out). n must be a multiple of 4.
// ---------------------------------------------------------------------------
__global__ __launch_bounds__(256) void cvt_f32_to_f16(
    const float* __restrict__ src, _Float16* __restrict__ dst, long n)
{
    const long i = ((long)blockIdx.x * 256 + threadIdx.x) * 4;
    if (i < n) {
        const v4f v = *(const v4f*)(src + i);
        v4h h;
        for (int j = 0; j < 4; ++j) h[j] = (_Float16)v[j];
        *(v4h*)(dst + i) = h;
    }
}

// ---------------------------------------------------------------------------
// Weight transpose + convert: Wt[n][k] = (f16)W[k][n].  K = 256 fixed.
// One block per output row n; threads sweep k (coalesced 2B writes).
// ---------------------------------------------------------------------------
__global__ __launch_bounds__(256) void transpose_w(
    const float* __restrict__ W, _Float16* __restrict__ Wt, int Ncols)
{
    const int n = blockIdx.x;
    const int k = threadIdx.x;
    Wt[(long)n * 256 + k] = (_Float16)W[(long)k * Ncols + n];
}

// ---------------------------------------------------------------------------
// Cache-direct f16 GEMM, K = 256:  out[M x N] = Ah[M x 256] * Wt[N x 256]^T + bias
// Block = 256 threads = 8 waves; block tile 128 rows x 64 cols.
// Wave w owns rows [w*16, w*16+16) and all 64 cols: 4 accumulators sharing
// one A fragment per K-step. No LDS, no barriers; EXEC stays all-1s for WMMA
// (OOB rows are address-clamped, only the stores are guarded).
// Fragment layouts used here (A and B share one k order):
//   A 16x32: lane(0-15)=row with K[0..15]; lanes 16-31 K[16..31] (contiguous)
//   B 32x16: lane(0-15)=col with K[0..15]; lanes 16-31 K[16..31] (contiguous)
// ---------------------------------------------------------------------------
template <bool OUT_F16>
__global__ __launch_bounds__(256) void gemm_k256_h(
    const _Float16* __restrict__ Ah,  // [Mrows x 256] f16
    const _Float16* __restrict__ Bt,  // [Ncols x 256] f16 (transposed weight)
    const float* __restrict__ bias,   // [Ncols]
    int Mrows, int Ncols,
    float* __restrict__ outF, _Float16* __restrict__ outH)
{
    const int tid  = threadIdx.x;
    const int lane = tid & 31;
    const int wv   = tid >> 5;

    const int rowbase = blockIdx.x * 128 + wv * 16;
    const int colbase = blockIdx.y * 64;

    // address-clamped A row (keeps EXEC uniform; OOB rows never stored)
    int arow = rowbase + (lane & 15);
    if (arow >= Mrows) arow = Mrows - 1;
    const _Float16* aptr = Ah + (long)arow * 256;
    const int kh = (lane >> 4) * 16;   // A half-wave K offset
    const int kb = (lane >> 4) * 16;   // B half-wave K offset

    const _Float16* bptr[4];
    for (int t = 0; t < 4; ++t)
        bptr[t] = Bt + (long)(colbase + t * 16 + (lane & 15)) * 256;

    v8f acc0 = {}, acc1 = {}, acc2 = {}, acc3 = {};

#pragma unroll
    for (int k0 = 0; k0 < 256; k0 += 32) {
        v16h a;
        {
            const v8h lo = *(const v8h*)(aptr + k0 + kh);
            const v8h hi = *(const v8h*)(aptr + k0 + kh + 8);
            for (int i = 0; i < 8; ++i) { a[i] = lo[i]; a[i + 8] = hi[i]; }
        }
        v16h b[4];
        for (int t = 0; t < 4; ++t) {
            const v8h lo = *(const v8h*)(bptr[t] + k0 + kb);
            const v8h hi = *(const v8h*)(bptr[t] + k0 + kb + 8);
            for (int i = 0; i < 8; ++i) { b[t][i] = lo[i]; b[t][i + 8] = hi[i]; }
        }
        acc0 = __builtin_amdgcn_wmma_f32_16x16x32_f16(false, a, false, b[0], (short)0, acc0, false, false);
        acc1 = __builtin_amdgcn_wmma_f32_16x16x32_f16(false, a, false, b[1], (short)0, acc1, false, false);
        acc2 = __builtin_amdgcn_wmma_f32_16x16x32_f16(false, a, false, b[2], (short)0, acc2, false, false);
        acc3 = __builtin_amdgcn_wmma_f32_16x16x32_f16(false, a, false, b[3], (short)0, acc3, false, false);
    }

    // store: VGPR r -> row r (lanes 0-15) / row 8+r (lanes 16-31); col = lane&15
    const int rofs = (lane >> 4) << 3;
    v8f accs[4] = {acc0, acc1, acc2, acc3};
    for (int t = 0; t < 4; ++t) {
        const int cb = colbase + t * 16 + (lane & 15);
        const float bi = bias[cb];
        for (int r = 0; r < 8; ++r) {
            const int gr = rowbase + rofs + r;
            if (gr < Mrows) {
                const float v = accs[t][r] + bi;
                if constexpr (OUT_F16) outH[(long)gr * Ncols + cb] = (_Float16)v;
                else                   outF[(long)gr * Ncols + cb] = v;
            }
        }
    }
}

// ---------------------------------------------------------------------------
// Sampling: one wave per (n, q, head); lane = channel d. Softmax over 16
// logits, 16 bilinear taps, 4 coalesced 64B corner gathers each from f16 value.
// ---------------------------------------------------------------------------
__global__ __launch_bounds__(256) void msda_sample(
    const float* __restrict__ refpt,     // (N, Lq, L, 2)
    const float* __restrict__ off,       // (NTOT, 256) fp32
    const float* __restrict__ attn,      // (NTOT, 128) fp32 logits
    const _Float16* __restrict__ valH,   // (NTOT, 256) f16
    _Float16* __restrict__ preH)         // (NTOT, 256) f16
{
    const int gw   = (int)((blockIdx.x * blockDim.x + threadIdx.x) >> 5);
    const int lane = threadIdx.x & 31;
    if (gw >= NTOT * MH) return;

    const int m   = gw & 7;
    const int row = gw >> 3;
    const int n   = row / LQ;

    const int Hs[4] = {100, 50, 25, 13};
    const int Ws[4] = {100, 50, 25, 13};
    const int St[4] = {0, 10000, 12500, 13125};

    const float* offr = off  + (long)row * 256 + m * 32;  // (L,P,2)
    const float* awr  = attn + (long)row * 128 + m * 16;  // (L*P)

    float mx = awr[0];
    for (int i = 1; i < 16; ++i) mx = fmaxf(mx, awr[i]);
    float wgt[16], s = 0.f;
    for (int i = 0; i < 16; ++i) { wgt[i] = __expf(awr[i] - mx); s += wgt[i]; }
    const float inv = 1.f / s;

    float acc = 0.f;
    const long vch = (long)m * 32 + lane;

    for (int lvl = 0; lvl < 4; ++lvl) {
        const int H = Hs[lvl], W = Ws[lvl];
        const float rx = refpt[(long)row * 8 + lvl * 2 + 0];
        const float ry = refpt[(long)row * 8 + lvl * 2 + 1];
        const long lvbase = (long)n * LEN + St[lvl];
        for (int p = 0; p < 4; ++p) {
            const int lp = lvl * 4 + p;
            const float ox = offr[lp * 2 + 0];
            const float oy = offr[lp * 2 + 1];
            const float x = (rx + ox / (float)W) * (float)W - 0.5f;
            const float y = (ry + oy / (float)H) * (float)H - 0.5f;
            const float x0f = floorf(x), y0f = floorf(y);
            const float fx = x - x0f, fy = y - y0f;
            const int x0 = (int)x0f, y0 = (int)y0f;
            const float aw  = wgt[lp] * inv;
            const float w00 = (1.f - fx) * (1.f - fy) * aw;
            const float w10 = fx * (1.f - fy) * aw;
            const float w01 = (1.f - fx) * fy * aw;
            const float w11 = fx * fy * aw;

            if (y0 >= 0 && y0 < H) {
                const long rb = lvbase + (long)y0 * W;
                if (x0 >= 0 && x0 < W)
                    acc += w00 * (float)valH[(rb + x0) * 256 + vch];
                if (x0 + 1 >= 0 && x0 + 1 < W)
                    acc += w10 * (float)valH[(rb + x0 + 1) * 256 + vch];
            }
            if (y0 + 1 >= 0 && y0 + 1 < H) {
                const long rb = lvbase + (long)(y0 + 1) * W;
                if (x0 >= 0 && x0 < W)
                    acc += w01 * (float)valH[(rb + x0) * 256 + vch];
                if (x0 + 1 >= 0 && x0 + 1 < W)
                    acc += w11 * (float)valH[(rb + x0 + 1) * 256 + vch];
            }
        }
    }
    preH[(long)row * 256 + m * 32 + lane] = (_Float16)acc;
}

// ---------------------------------------------------------------------------
extern "C" void kernel_launch(void* const* d_in, const int* in_sizes, int n_in,
                              void* d_out, int out_size, void* d_ws, size_t ws_size,
                              hipStream_t stream) {
    const float* query = (const float*)d_in[0];
    const float* refpt = (const float*)d_in[1];
    const float* input = (const float*)d_in[2];
    // d_in[3] shapes, d_in[4] starts: static in reference -> compile-time consts
    const float* Wv    = (const float*)d_in[5];
    const float* bv    = (const float*)d_in[6];
    const float* Woff  = (const float*)d_in[7];
    const float* boff  = (const float*)d_in[8];
    const float* Wattn = (const float*)d_in[9];
    const float* battn = (const float*)d_in[10];
    const float* Wo    = (const float*)d_in[11];
    const float* bo    = (const float*)d_in[12];
    float* out = (float*)d_out;

    // workspace layout (~191 MB, L2-scale)
    char* ws = (char*)d_ws;
    size_t o = 0;
    _Float16* valH   = (_Float16*)(ws + o); o += (size_t)NTOT * 256 * 2;
    _Float16* queryH = (_Float16*)(ws + o); o += (size_t)NTOT * 256 * 2;
    _Float16* inputH = (_Float16*)(ws + o); o += (size_t)NTOT * 256 * 2;
    _Float16* preH   = (_Float16*)(ws + o); o += (size_t)NTOT * 256 * 2;
    float*    offF   = (float*)   (ws + o); o += (size_t)NTOT * 256 * 4;
    float*    attF   = (float*)   (ws + o); o += (size_t)NTOT * 128 * 4;
    _Float16* WtV    = (_Float16*)(ws + o); o += (size_t)256 * 256 * 2;
    _Float16* WtOff  = (_Float16*)(ws + o); o += (size_t)256 * 256 * 2;
    _Float16* WtA    = (_Float16*)(ws + o); o += (size_t)128 * 256 * 2;
    _Float16* WtOut  = (_Float16*)(ws + o);

    const dim3 blk(256);
    const long nElem = (long)NTOT * 256;                 // 13,613,056 (mult of 1024)
    const dim3 gCvt((unsigned)(nElem / 4 / 256));
    const dim3 gM2((NTOT + 127) / 128, 4);               // N = 256 (4 x 64-col strips)
    const dim3 gM1((NTOT + 127) / 128, 2);               // N = 128

    // 0) precision/layout staging
    cvt_f32_to_f16<<<gCvt, blk, 0, stream>>>(input, inputH, nElem);
    cvt_f32_to_f16<<<gCvt, blk, 0, stream>>>(query, queryH, nElem);
    transpose_w<<<dim3(256), blk, 0, stream>>>(Wv,    WtV,   256);
    transpose_w<<<dim3(256), blk, 0, stream>>>(Woff,  WtOff, 256);
    transpose_w<<<dim3(128), blk, 0, stream>>>(Wattn, WtA,   128);
    transpose_w<<<dim3(256), blk, 0, stream>>>(Wo,    WtOut, 256);

    // 1) value = input @ Wv + bv  -> f16
    gemm_k256_h<true ><<<gM2, blk, 0, stream>>>(inputH, WtV, bv, NTOT, 256, nullptr, valH);
    // 2) offsets = query @ W_off + b_off -> f32
    gemm_k256_h<false><<<gM2, blk, 0, stream>>>(queryH, WtOff, boff, NTOT, 256, offF, nullptr);
    // 3) attn logits = query @ W_attn + b_attn -> f32
    gemm_k256_h<false><<<gM1, blk, 0, stream>>>(queryH, WtA, battn, NTOT, 128, attF, nullptr);
    // 4) softmax + bilinear sampling + head accumulate -> f16
    msda_sample<<<dim3((NTOT * MH) / 8), blk, 0, stream>>>(refpt, offF, attF, valH, preH);
    // 5) out = pre @ Wo + bo -> fp32 into d_out
    gemm_k256_h<false><<<gM2, blk, 0, stream>>>(preH, WtOut, bo, NTOT, 256, out, nullptr);
}
